// MKMMDLoss_22093311771155
// MI455X (gfx1250) — matrix operand order repaired and checked by reading, not compiled, not run on hardware
//
#include <hip/hip_runtime.h>

typedef unsigned short u16;
typedef u16    v4u   __attribute__((ext_vector_type(4)));
typedef u16    v8u   __attribute__((ext_vector_type(8)));
typedef u16    v16u  __attribute__((ext_vector_type(16)));
typedef __bf16 v16bf __attribute__((ext_vector_type(16)));
typedef float  v8f   __attribute__((ext_vector_type(8)));
typedef unsigned int u32x4 __attribute__((ext_vector_type(4)));
typedef int          i32x4 __attribute__((ext_vector_type(4)));
typedef int          i32x8 __attribute__((ext_vector_type(8)));

#define BN     4096        // batch size (rows of xs / xt)
#define KD     256         // feature dim
#define TILE_M 64          // output tile rows per workgroup
#define TILE_N 128         // output tile cols per workgroup
#define KC     64          // bf16 K elements staged per chunk
#define NCHUNK (KD / KC)   // 4
#define LDRB   (KC + 8)    // bf16 elements per LDS plane row (16B padded)

// LDS layout (u16 element offsets within one buffer)
#define PLA_E  (TILE_M * LDRB)            // 4608  (A plane)
#define PLB_E  (TILE_N * LDRB)            // 9216  (B plane)
#define AHI_E  0
#define ALO_E  PLA_E
#define BHI_E  (2 * PLA_E)
#define BLO_E  (2 * PLA_E + PLB_E)
#define BUF_E  (2 * PLA_E + 2 * PLB_E)    // 27648 u16 = 55296 B per buffer
#define SMEM_NORM_OFF (2 * BUF_E * 2)     // 110592 B: norms (f32) region
#define SMEM_SUM_OFF  (SMEM_NORM_OFF + (TILE_M + TILE_N) * 4)
#define SMEM_TOTAL    (SMEM_SUM_OFF + 16)

#if defined(__has_builtin)
#  if __has_builtin(__builtin_amdgcn_tensor_load_to_lds)
#    define USE_TDM 1
#  endif
#endif
#ifndef USE_TDM
#  define USE_TDM 0
#endif

// ---------------------------------------------------------------------------
// ws layout:
//  float [0..2]   pass-1 distance sums (xx, yy, xy)
//  float [3..5]   coefficients 1/(2*mean)
//  float [6..8]   pass-2 kernel sums (xx, yy, xy)
//  float [16 .. 16+8191]  row norms (xs then xt)
//  u16 planes after that: XsHi, XsLo, XtHi, XtLo, each BN*KD (8 MB total)
// ---------------------------------------------------------------------------

__global__ void mk_zero_ws(float* __restrict__ ws) {
    if (threadIdx.x < 16) ws[threadIdx.x] = 0.f;
}

__global__ void mk_row_norms(const float* __restrict__ xs,
                             const float* __restrict__ xt,
                             float* __restrict__ nxs,
                             float* __restrict__ nxt) {
    const int lane = threadIdx.x & 31;
    const int wave = threadIdx.x >> 5;
    const int row  = blockIdx.x * 8 + wave;
    const float* __restrict__ src = (blockIdx.y == 0) ? xs : xt;
    float* __restrict__ dst       = (blockIdx.y == 0) ? nxs : nxt;
    const float* p = src + (size_t)row * KD;
    float s = 0.f;
#pragma unroll
    for (int j = 0; j < KD; j += 32) {
        float v = p[j + lane];
        s += v * v;
    }
#pragma unroll
    for (int off = 16; off >= 1; off >>= 1)
        s += __shfl_xor(s, off, 32);
    if (lane == 0) dst[row] = s;
}

// f32 -> bf16 hi (truncated top 16 bits) + bf16 lo (truncated residual).
__device__ __forceinline__ void split_store(u16* __restrict__ hi,
                                            u16* __restrict__ lo,
                                            int idx, float4 f) {
    const float fx[4] = {f.x, f.y, f.z, f.w};
    v4u h, l;
#pragma unroll
    for (int i = 0; i < 4; ++i) {
        const unsigned u  = __float_as_uint(fx[i]);
        const unsigned hu = u & 0xffff0000u;
        h[i] = (u16)(u >> 16);
        const float r = fx[i] - __uint_as_float(hu);
        l[i] = (u16)(__float_as_uint(r) >> 16);
    }
    *(v4u*)(hi + idx) = h;
    *(v4u*)(lo + idx) = l;
}

// One-time global bf16-split of the inputs.
__global__ void mk_split(const float* __restrict__ xs,
                         const float* __restrict__ xt,
                         u16* __restrict__ planes) {
    const float* x = blockIdx.y ? xt : xs;
    u16* hi = planes + (size_t)blockIdx.y * 2 * BN * KD;
    u16* lo = hi + (size_t)BN * KD;
    const int idx = (blockIdx.x * 256 + threadIdx.x) * 4;
    const float4 f = *(const float4*)(x + idx);
    split_store(hi, lo, idx, f);
}

// Two contiguous 16B runs from an LDS bf16 plane -> one WMMA operand.
__device__ __forceinline__ v16bf load_frag(const u16* p0, const u16* p1) {
    const v8u a = *(const v8u*)p0;
    const v8u b = *(const v8u*)p1;
    const v16u c = __builtin_shufflevector(a, b, 0, 1, 2, 3, 4, 5, 6, 7,
                                                 8, 9, 10, 11, 12, 13, 14, 15);
    return __builtin_bit_cast(v16bf, c);
}

#define WMMA_BF16(A, B, C) \
    __builtin_amdgcn_wmma_f32_16x16x32_bf16(false, (A), false, (B), (short)0, (C), false, false)

#if USE_TDM && defined(__HIP_DEVICE_COMPILE__)
// Issue one 2D TDM tile load: rows x KC bf16 elements, tensor stride KD,
// LDS rows padded by 16B every 128B (pad_interval=32 dwords, pad_amount=4).
__device__ __forceinline__ void tdm_load(unsigned lds_byte_off,
                                         const u16* __restrict__ gsrc,
                                         int rows) {
    const unsigned long long ga = (unsigned long long)(const void*)gsrc;
    u32x4 g0;
    g0[0] = 1u;                                        // count=1, user mode
    g0[1] = lds_byte_off;                              // lds_addr
    g0[2] = (unsigned)ga;                              // global_addr[31:0]
    g0[3] = (unsigned)((ga >> 32) & 0x01ffffffull) | (2u << 30);  // type=2
    i32x8 g1;
    g1[0] = (1 << 16)        // data_size = 2 bytes
          | (1 << 20)        // pad_enable
          | (4 << 22)        // pad_interval: 32 dwords (one 128B row)
          | (3 << 25);       // pad_amount: 4 dwords (16B)
    g1[1] = (int)((unsigned)(KD & 0xffff) << 16);      // tensor_dim0 lo16
    g1[2] = (int)((unsigned)(BN & 0xffff) << 16);      // tensor_dim1 lo16
    g1[3] = (int)((unsigned)KC << 16);                 // tile_dim0
    g1[4] = rows;                                      // tile_dim1
    g1[5] = KD;                                        // tensor_dim0_stride
    g1[6] = 0;
    g1[7] = 0;
    const i32x4 gz4 = {0, 0, 0, 0};
    const i32x8 gz8 = {0, 0, 0, 0, 0, 0, 0, 0};
    __builtin_amdgcn_tensor_load_to_lds(g0, g1, gz4, gz4, gz8, 0);
}
#endif

// Tiled distance / kernel-sum pass: bf16-split GEMM on the matrix pipe,
// TDM-staged double-buffered LDS tiles.
// PASS==1: accumulate sum of distances; PASS==2: sum of 5 Gaussian kernels.
template <int PASS>
__global__ void mk_mmd_tiles(const u16* __restrict__ planes,
                             const float* __restrict__ nxs,
                             const float* __restrict__ nxt,
                             float* __restrict__ acc,     // 3 accumulators
                             const float* __restrict__ coef) {
    const int z  = blockIdx.z;               // 0: xx, 1: yy, 2: xy
    const int bx = blockIdx.x, by = blockIdx.y;
    // dxx/dyy symmetric: drop tiles fully below the diagonal.
    if (z < 2 && bx >= 2 * by + 2) return;

    extern __shared__ __align__(16) char smem[];
    u16*   lds    = (u16*)smem;
    float* shNorm = (float*)(smem + SMEM_NORM_OFF);
    float* sSum   = (float*)(smem + SMEM_SUM_OFF);

    const int tid  = threadIdx.x;
    const int lane = tid & 31;
    const int wave = tid >> 5;

    const u16* XsHi = planes;
    const u16* XtHi = planes + (size_t)2 * BN * KD;
    const u16 *XH, *YH; const float *XN, *YN;
    if (z == 0)      { XH = XsHi; YH = XsHi; XN = nxs; YN = nxs; }
    else if (z == 1) { XH = XtHi; YH = XtHi; XN = nxt; YN = nxt; }
    else             { XH = XsHi; YH = XtHi; XN = nxs; YN = nxt; }
    const u16* XL = XH + (size_t)BN * KD;
    const u16* YL = YH + (size_t)BN * KD;

    const int i0 = bx * TILE_M;
    const int j0 = by * TILE_N;

    if (tid == 0) sSum[0] = 0.f;
    if (tid < TILE_M)                 shNorm[tid] = XN[i0 + tid];
    else if (tid < TILE_M + TILE_N)   shNorm[tid] = YN[j0 + (tid - TILE_M)];

    // 8 waves -> 4x8 grid of 16x16 subtiles; each wave: 1 row x 4 cols.
    const int sr     = wave & 3;
    const int scbase = (wave >> 2) * 4;
    const int half   = lane >> 4;
    const int mrow   = lane & 15;

    const int amo = (sr * 16 + mrow) * LDRB + half * 16;  // A frag: +k0, +8
    int bno[4];
#pragma unroll
    for (int t = 0; t < 4; ++t)                            // B frag: +k0, +8
        bno[t] = ((scbase + t) * 16 + mrow) * LDRB + half * 16;

    v8f cacc[4];
#pragma unroll
    for (int t = 0; t < 4; ++t) cacc[t] = (v8f){0.f,0.f,0.f,0.f,0.f,0.f,0.f,0.f};

#if USE_TDM && defined(__HIP_DEVICE_COMPILE__)
    if (wave == 0) {
        tdm_load(AHI_E * 2, XH + (size_t)i0 * KD, TILE_M);
        tdm_load(ALO_E * 2, XL + (size_t)i0 * KD, TILE_M);
        tdm_load(BHI_E * 2, YH + (size_t)j0 * KD, TILE_N);
        tdm_load(BLO_E * 2, YL + (size_t)j0 * KD, TILE_N);
    }
#endif

    for (int c = 0; c < NCHUNK; ++c) {
        const int kt = c * KC;
        (void)kt;
#if USE_TDM && defined(__HIP_DEVICE_COMPILE__)
        const unsigned bb = (unsigned)(c & 1) * BUF_E;
        if (wave == 0) {
            if (c + 1 < NCHUNK) {
                const unsigned nb = (unsigned)((c + 1) & 1) * BUF_E * 2;
                const int nk = (c + 1) * KC;
                tdm_load(nb + AHI_E * 2, XH + (size_t)i0 * KD + nk, TILE_M);
                tdm_load(nb + ALO_E * 2, XL + (size_t)i0 * KD + nk, TILE_M);
                tdm_load(nb + BHI_E * 2, YH + (size_t)j0 * KD + nk, TILE_N);
                tdm_load(nb + BLO_E * 2, YL + (size_t)j0 * KD + nk, TILE_N);
                __builtin_amdgcn_s_wait_tensorcnt(4);  // chunk c complete
            } else {
                __builtin_amdgcn_s_wait_tensorcnt(0);  // last chunk complete
            }
        }
        __syncthreads();
#else
        const unsigned bb = 0;
        // Fallback staging: 384 plane-rows x 8 b128 slots = 3072 copies.
#pragma unroll
        for (int v = 0; v < 12; ++v) {
            const int lin = tid + v * 256;
            const int row = lin >> 3;
            const int s8  = (lin & 7) * 8;
            const u16* src; unsigned dstE;
            if (row < 64)       { src = XH + (size_t)(i0 + row) * KD;       dstE = AHI_E + row * LDRB; }
            else if (row < 128) { src = XL + (size_t)(i0 + row - 64) * KD;  dstE = ALO_E + (row - 64) * LDRB; }
            else if (row < 256) { src = YH + (size_t)(j0 + row - 128) * KD; dstE = BHI_E + (row - 128) * LDRB; }
            else                { src = YL + (size_t)(j0 + row - 256) * KD; dstE = BLO_E + (row - 256) * LDRB; }
            *(v8u*)(lds + dstE + s8) = *(const v8u*)(src + kt + s8);
        }
        __syncthreads();
#endif
        const u16* Ah = lds + bb + AHI_E;
        const u16* Al = lds + bb + ALO_E;
        const u16* Bh = lds + bb + BHI_E;
        const u16* Bl = lds + bb + BLO_E;
#pragma unroll
        for (int k0 = 0; k0 < KC; k0 += 32) {
            const v16bf ah = load_frag(Ah + amo + k0, Ah + amo + k0 + 8);
            const v16bf al = load_frag(Al + amo + k0, Al + amo + k0 + 8);
#pragma unroll
            for (int t = 0; t < 4; ++t) {
                const v16bf bh = load_frag(Bh + bno[t] + k0, Bh + bno[t] + k0 + 8);
                const v16bf bl = load_frag(Bl + bno[t] + k0, Bl + bno[t] + k0 + 8);
                // dot = hi*hi + hi*lo + lo*hi   (lo*lo ~ 2^-16: dropped)
                cacc[t] = WMMA_BF16(ah, bh, cacc[t]);
                cacc[t] = WMMA_BF16(ah, bl, cacc[t]);
                cacc[t] = WMMA_BF16(al, bh, cacc[t]);
            }
        }
        __syncthreads();
    }

    // Epilogue: distances (and kernels) from dot products + norms.
    float cf = 0.f;
    if (PASS == 2) cf = coef[z];
    float local = 0.f;

#pragma unroll
    for (int t = 0; t < 4; ++t) {
#pragma unroll
        for (int r8 = 0; r8 < 8; ++r8) {
            const int M  = r8 + (half << 3);
            const int gi = i0 + sr * 16 + M;
            const int gj = j0 + (scbase + t) * 16 + (lane & 15);
            float wv = 1.f;
            if (z < 2) wv = (gj > gi) ? 2.f : ((gj == gi) ? 1.f : 0.f);
            const float dot = cacc[t][r8];
            float sq = shNorm[sr * 16 + M]
                     + shNorm[TILE_M + (scbase + t) * 16 + (lane & 15)]
                     - 2.f * dot;
            float d = (sq > 0.f) ? sqrtf(sq) : 0.f;   // cdist zero-dist rule
            if (PASS == 1) {
                local += wv * d;
            } else {
                // alphas = 2^-3..2^1  =>  e^8 + e^4 + e^2 + e + sqrt(e)
                const float e1 = __expf(-d * cf);
                const float e2 = e1 * e1;
                const float e4 = e2 * e2;
                const float e8 = e4 * e4;
                local += wv * (e8 + e4 + e2 + e1 + sqrtf(e1));
            }
        }
    }

#pragma unroll
    for (int off = 16; off >= 1; off >>= 1)
        local += __shfl_xor(local, off, 32);
    if (lane == 0) atomicAdd(&sSum[0], local);
    __syncthreads();
    if (tid == 0) atomicAdd(&acc[z], sSum[0]);
}

__global__ void mk_coef(float* __restrict__ ws) {
    const int t = threadIdx.x;
    if (t < 3) {
        // mean = sum / B^2 ;  coef = 1/(2*mean) = B^2 / (2*sum)
        ws[3 + t] = ((float)BN * (float)BN * 0.5f) / ws[t];
    }
}

__global__ void mk_final(const float* __restrict__ ws, float* __restrict__ out) {
    const float S = ws[6] + ws[7] - 2.f * ws[8];
    const float v = S / ((float)BN * ((float)BN - 1.f));
    float l = sqrtf(v);
    if (!(l == l)) l = 0.f;   // NaN -> 0 per reference
    out[0] = l;
}

extern "C" void kernel_launch(void* const* d_in, const int* in_sizes, int n_in,
                              void* d_out, int out_size, void* d_ws, size_t ws_size,
                              hipStream_t stream) {
    const float* xs = (const float*)d_in[0];
    const float* xt = (const float*)d_in[1];
    float* ws     = (float*)d_ws;
    float* nxs    = ws + 16;
    float* nxt    = ws + 16 + BN;
    u16*   planes = (u16*)(ws + 16 + 2 * BN);   // 4 * BN*KD bf16 = 8 MB
    float* out    = (float*)d_out;

    mk_zero_ws<<<1, 16, 0, stream>>>(ws);
    mk_row_norms<<<dim3(BN / 8, 2), 256, 0, stream>>>(xs, xt, nxs, nxt);
    mk_split<<<dim3(BN * KD / 1024, 2), 256, 0, stream>>>(xs, xt, planes);

    dim3 grid(BN / TILE_M, BN / TILE_N, 3);
    mk_mmd_tiles<1><<<grid, 256, SMEM_TOTAL, stream>>>(planes, nxs, nxt, ws + 0, ws + 3);
    mk_coef<<<1, 32, 0, stream>>>(ws);
    mk_mmd_tiles<2><<<grid, 256, SMEM_TOTAL, stream>>>(planes, nxs, nxt, ws + 6, ws + 3);
    mk_final<<<1, 1, 0, stream>>>(ws, out);
}
